// MaskedMultiHeadAttention_33457795236349
// MI455X (gfx1250) — matrix operand order repaired and checked by reading, not compiled, not run on hardware
//
#include <hip/hip_runtime.h>
#include <stdint.h>
#include <stddef.h>

// ---------------------------------------------------------------------------
// Types for CDNA5 WMMA (wave32): v_wmma_f32_16x16x32_bf16
// ---------------------------------------------------------------------------
typedef __bf16 bf16;
typedef __attribute__((ext_vector_type(16))) __bf16   v16bf;
typedef __attribute__((ext_vector_type(4)))  __bf16   v4bf;
typedef __attribute__((ext_vector_type(8)))  float    v8f;
typedef __attribute__((ext_vector_type(4)))  uint32_t u32x4;

union FragBF { v16bf v; u32x4 q[2]; };

#define WMMA_BF16(a, b, c) \
  __builtin_amdgcn_wmma_f32_16x16x32_bf16(false, (a), false, (b), (short)0, (c), false, false)

__device__ __forceinline__ v8f vzero8() {
  v8f z;
#pragma unroll
  for (int i = 0; i < 8; ++i) z[i] = 0.0f;
  return z;
}

// A-fragment (16x32 bf16, M x K), row-major source with stride ld (elements).
// Lane l: row m = l&15, half g = l>>4. V0..V3 hold K = 16g..16g+7 (16B),
// V4..V7 hold K = 16g+8..16g+15 (16B)  -> two 16-byte loads per lane.
__device__ __forceinline__ v16bf load_a_frag(const bf16* p, int ld, int lane) {
  const int m = lane & 15, g = (lane >> 4) & 1;
  FragBF f;
  f.q[0] = *(const u32x4*)(p + (size_t)m * ld + 16 * g);
  f.q[1] = *(const u32x4*)(p + (size_t)m * ld + 16 * g + 8);
  return f.v;
}

// B-fragment (32x16 bf16, K x N) where B[k][n] = W[n][k] (W row-major, stride
// ld). Lane l: col n = l&15, half g = l>>4, V0..V7 hold K = 16g..16g+15
// -> 32 contiguous bytes in W row n -> two 16-byte loads per lane.
__device__ __forceinline__ v16bf load_bT_frag(const bf16* w, int ld, int lane) {
  const int n = lane & 15, g = (lane >> 4) & 1;
  const bf16* p = w + (size_t)n * ld + 16 * g;
  FragBF f;
  f.q[0] = *(const u32x4*)(p);
  f.q[1] = *(const u32x4*)(p + 8);
  return f.v;
}

// ---------------------------------------------------------------------------
// Problem constants (fixed by the reference)
// ---------------------------------------------------------------------------
#define BB   8
#define SS   1024
#define DD   768
#define HH   12
#define HD   64
#define MM   (BB * SS)          // 8192 tokens
#define LOG2E 1.4426950408889634f
#define SCALE_Q (0.125f * LOG2E)   // hd^-0.5 folded with log2(e): exp2 softmax
#define NEG_BIG -1.0e9f

// ---------------------------------------------------------------------------
// fp32 -> bf16 elementwise convert (4 elements / thread)
// ---------------------------------------------------------------------------
__global__ void f32_to_bf16_kernel(const float* __restrict__ src,
                                   bf16* __restrict__ dst, int n4) {
  int i = blockIdx.x * blockDim.x + threadIdx.x;
  const int stride = gridDim.x * blockDim.x;
  for (; i < n4; i += stride) {
    const float4 f = ((const float4*)src)[i];
    v4bf o;
    o.x = (bf16)f.x; o.y = (bf16)f.y; o.z = (bf16)f.z; o.w = (bf16)f.w;
    ((v4bf*)dst)[i] = o;
  }
}

// ---------------------------------------------------------------------------
// QKV projection: C[M, 768] = Xbf @ W^T + b, per-token mask, Q scaled.
//  blockIdx.z: 0 -> Q [B,H,S,hd], 1 -> K [B,H,S,hd], 2 -> V transposed [B,H,hd,S]
//  block = 128 threads (4 waves); wave computes a 16(M) x 64(N) tile.
//  grid = (MM/64, HH, 3); blockIdx.y == head (64 cols == one head).
//  Fully-unrolled, double-buffered K-loop: next-step loads are issued before
//  current WMMAs (partial loadcnt waits), and unrolling lets SSA renaming
//  eliminate the ping-pong register copies.
// ---------------------------------------------------------------------------
__global__ __launch_bounds__(128)
void qkv_proj_kernel(const bf16* __restrict__ X,
                     const bf16* __restrict__ Wq, const bf16* __restrict__ Wk,
                     const bf16* __restrict__ Wv,
                     const float* __restrict__ bq, const float* __restrict__ bk,
                     const float* __restrict__ bv,
                     const unsigned char* __restrict__ mask,  // [B*S] bool8
                     bf16* __restrict__ Q, bf16* __restrict__ K,
                     bf16* __restrict__ Vt) {
  const int lane = threadIdx.x & 31;
  const int wv_id = threadIdx.x >> 5;
  const int m0 = (blockIdx.x * 4 + wv_id) * 16;
  const int h = blockIdx.y;
  const int n0 = h * HD;
  const int which = blockIdx.z;

  const bf16* W = (which == 0) ? Wq : (which == 1) ? Wk : Wv;
  const float* bias = (which == 0) ? bq : (which == 1) ? bk : bv;

  const bf16* Ap = X + (size_t)m0 * DD;
  const bf16* Wp = W + (size_t)n0 * DD;

  v8f acc[4];
#pragma unroll
  for (int j = 0; j < 4; ++j) acc[j] = vzero8();

  // prime the pipeline (k0 = 0)
  v16bf a = load_a_frag(Ap, DD, lane);
  v16bf bfr[4];
#pragma unroll
  for (int j = 0; j < 4; ++j)
    bfr[j] = load_bT_frag(Wp + (size_t)(16 * j) * DD, DD, lane);

#pragma unroll
  for (int k0 = 0; k0 < DD - 32; k0 += 32) {
    const int kn = k0 + 32;
    __builtin_prefetch(Ap + kn + 32, 0, 3);
    const v16bf a_n = load_a_frag(Ap + kn, DD, lane);
    v16bf b_n[4];
#pragma unroll
    for (int j = 0; j < 4; ++j)
      b_n[j] = load_bT_frag(Wp + (size_t)(16 * j) * DD + kn, DD, lane);
#pragma unroll
    for (int j = 0; j < 4; ++j) acc[j] = WMMA_BF16(a, bfr[j], acc[j]);
    a = a_n;
#pragma unroll
    for (int j = 0; j < 4; ++j) bfr[j] = b_n[j];
  }
  // final K tile (no further loads)
#pragma unroll
  for (int j = 0; j < 4; ++j) acc[j] = WMMA_BF16(a, bfr[j], acc[j]);

  const int n = lane & 15, g = (lane >> 4) & 1;
  const float sc = (which == 0) ? SCALE_Q : 1.0f;
#pragma unroll
  for (int j = 0; j < 4; ++j) {
    const int d = 16 * j + n;                 // within-head feature index
    const float bj = bias[n0 + d];
#pragma unroll
    for (int r = 0; r < 8; ++r) {
      const int t = m0 + r + 8 * g;           // global token
      const float mk = mask[t] ? 1.0f : 0.0f;
      const float val = (acc[j][r] + bj) * mk * sc;
      const int bidx = t >> 10, s = t & (SS - 1);
      const size_t head = (size_t)bidx * HH + h;
      if (which == 2)
        Vt[(head * HD + d) * SS + s] = (bf16)val;          // [B,H,hd,S]
      else {
        bf16* dst = (which == 0) ? Q : K;
        dst[(head * SS + s) * HD + d] = (bf16)val;         // [B,H,S,hd]
      }
    }
  }
}

// ---------------------------------------------------------------------------
// Fused flash attention. One wave = 16 query rows x hd=64 output.
//  block = 256 (8 waves -> 128 q rows); grid = (S/128, H, B).
//  Per 32-key tile: V-frags issued first (consumed last), next K-frags
//  issued before QK^T WMMAs -> softmax VALU overlaps in-flight loads.
//  Softmax runs in exp2 domain (log2e folded into Q scale). Key loop
//  unrolled x2 so the K-fragment ping-pong copies vanish.
// ---------------------------------------------------------------------------
__global__ __launch_bounds__(256)
void attn_kernel(const bf16* __restrict__ Q, const bf16* __restrict__ K,
                 const bf16* __restrict__ Vt,
                 const unsigned char* __restrict__ mask,
                 bf16* __restrict__ AO) {
  __shared__ bf16 lds_p[8][16 * 32];          // per-wave P scratch (16x32)
  const int lane = threadIdx.x & 31;
  const int wv_id = threadIdx.x >> 5;
  const int h = blockIdx.y, b = blockIdx.z;
  const int q0 = blockIdx.x * 128 + wv_id * 16;
  const size_t head = (size_t)b * HH + h;
  const bf16* Qh = Q + (head * SS + q0) * HD;
  const bf16* Kh = K + head * SS * HD;
  const bf16* Vh = Vt + head * HD * SS;
  const unsigned char* mb = mask + (size_t)b * SS;

  const v16bf aq0 = load_a_frag(Qh, HD, lane);        // hd 0..31
  const v16bf aq1 = load_a_frag(Qh + 32, HD, lane);   // hd 32..63

  v8f o[4];
#pragma unroll
  for (int j = 0; j < 4; ++j) o[j] = vzero8();
  float mx[8], ls[8];
#pragma unroll
  for (int r = 0; r < 8; ++r) { mx[r] = -3.0e38f; ls[r] = 0.0f; }

  const int n = lane & 15, g = (lane >> 4) & 1;
  bf16* pw = &lds_p[wv_id][0];

  // prime K-fragment pipeline (kt = 0)
  v16bf kb0 = load_bT_frag(Kh, HD, lane);
  v16bf kb1 = load_bT_frag(Kh + 32, HD, lane);
  v16bf kb2 = load_bT_frag(Kh + 16 * HD, HD, lane);
  v16bf kb3 = load_bT_frag(Kh + 16 * HD + 32, HD, lane);

#pragma unroll 2
  for (int kt = 0; kt < SS; kt += 32) {
    // V fragments for this tile: issued first, consumed last
    v16bf vb[4];
#pragma unroll
    for (int j = 0; j < 4; ++j)
      vb[j] = load_bT_frag(Vh + (size_t)(16 * j) * SS + kt, SS, lane);

    // next K tile fragments (wrap on last iteration; values unused)
    const int ktn = (kt + 32 < SS) ? kt + 32 : 0;
    const bf16* Ktn = Kh + (size_t)ktn * HD;
    const v16bf kn0 = load_bT_frag(Ktn, HD, lane);
    const v16bf kn1 = load_bT_frag(Ktn + 32, HD, lane);
    const v16bf kn2 = load_bT_frag(Ktn + 16 * HD, HD, lane);
    const v16bf kn3 = load_bT_frag(Ktn + 16 * HD + 32, HD, lane);

    v8f s0 = vzero8(), s1 = vzero8();
    s0 = WMMA_BF16(aq0, kb0, s0);
    s0 = WMMA_BF16(aq1, kb1, s0);
    s1 = WMMA_BF16(aq0, kb2, s1);
    s1 = WMMA_BF16(aq1, kb3, s1);

    // additive key mask (n is this lane's key column within the tile)
    const float mk0 = mb[kt + n] ? 0.0f : NEG_BIG;
    const float mk1 = mb[kt + 16 + n] ? 0.0f : NEG_BIG;
    v8f tmax;
#pragma unroll
    for (int r = 0; r < 8; ++r) {
      s0[r] += mk0; s1[r] += mk1;
      tmax[r] = fmaxf(s0[r], s1[r]);
    }
    // row max across the 16 lanes holding this row
#pragma unroll
    for (int d = 1; d < 16; d <<= 1) {
#pragma unroll
      for (int r = 0; r < 8; ++r)
        tmax[r] = fmaxf(tmax[r], __shfl_xor(tmax[r], d, 32));
    }
    v8f psum;
#pragma unroll
    for (int r = 0; r < 8; ++r) {
      const float mn = fmaxf(mx[r], tmax[r]);
      const float scl = __builtin_amdgcn_exp2f(mx[r] - mn);   // v_exp_f32
      mx[r] = mn;
      const float p0 = __builtin_amdgcn_exp2f(s0[r] - mn);
      const float p1 = __builtin_amdgcn_exp2f(s1[r] - mn);
      s0[r] = p0; s1[r] = p1;
      psum[r] = p0 + p1;
      ls[r] *= scl;
#pragma unroll
      for (int j = 0; j < 4; ++j) o[j][r] *= scl;
    }
#pragma unroll
    for (int d = 1; d < 16; d <<= 1) {
#pragma unroll
      for (int r = 0; r < 8; ++r) psum[r] += __shfl_xor(psum[r], d, 32);
    }
#pragma unroll
    for (int r = 0; r < 8; ++r) ls[r] += psum[r];

    // stage P (16 q-rows x 32 keys) into LDS row-major [m][k]; same-wave
    // LDS ops are in-order so no barrier is required.
#pragma unroll
    for (int r = 0; r < 8; ++r) {
      pw[(r + 8 * g) * 32 + n]      = (bf16)s0[r];
      pw[(r + 8 * g) * 32 + 16 + n] = (bf16)s1[r];
    }
    const v16bf pa = load_a_frag(pw, 32, lane);   // A-frag of P (ds loads)

#pragma unroll
    for (int j = 0; j < 4; ++j) o[j] = WMMA_BF16(pa, vb[j], o[j]);

    kb0 = kn0; kb1 = kn1; kb2 = kn2; kb3 = kn3;
  }

  // epilogue: normalize, apply query mask, store bf16 AO[token, D]
#pragma unroll
  for (int r = 0; r < 8; ++r) {
    const int s = q0 + r + 8 * g;
    const float qm = mb[s] ? 1.0f : 0.0f;
    const float inv = qm * __builtin_amdgcn_rcpf(ls[r]);
    bf16* dst = AO + ((size_t)b * SS + s) * DD + h * HD;
#pragma unroll
    for (int j = 0; j < 4; ++j) dst[16 * j + n] = (bf16)(o[j][r] * inv);
  }
}

// ---------------------------------------------------------------------------
// Output projection: out[M,768] = AO @ Wo^T + bo (fp32 out), fully unrolled
// double-buffered K-loop.
// ---------------------------------------------------------------------------
__global__ __launch_bounds__(128)
void out_proj_kernel(const bf16* __restrict__ AO, const bf16* __restrict__ Wo,
                     const float* __restrict__ bo, float* __restrict__ out) {
  const int lane = threadIdx.x & 31;
  const int wv_id = threadIdx.x >> 5;
  const int m0 = (blockIdx.x * 4 + wv_id) * 16;
  const int n0 = blockIdx.y * 64;

  const bf16* Ap = AO + (size_t)m0 * DD;
  const bf16* Wp = Wo + (size_t)n0 * DD;

  v8f acc[4];
#pragma unroll
  for (int j = 0; j < 4; ++j) acc[j] = vzero8();

  v16bf a = load_a_frag(Ap, DD, lane);
  v16bf bfr[4];
#pragma unroll
  for (int j = 0; j < 4; ++j)
    bfr[j] = load_bT_frag(Wp + (size_t)(16 * j) * DD, DD, lane);

#pragma unroll
  for (int k0 = 0; k0 < DD - 32; k0 += 32) {
    const int kn = k0 + 32;
    __builtin_prefetch(Ap + kn + 32, 0, 3);
    const v16bf a_n = load_a_frag(Ap + kn, DD, lane);
    v16bf b_n[4];
#pragma unroll
    for (int j = 0; j < 4; ++j)
      b_n[j] = load_bT_frag(Wp + (size_t)(16 * j) * DD + kn, DD, lane);
#pragma unroll
    for (int j = 0; j < 4; ++j) acc[j] = WMMA_BF16(a, bfr[j], acc[j]);
    a = a_n;
#pragma unroll
    for (int j = 0; j < 4; ++j) bfr[j] = b_n[j];
  }
#pragma unroll
  for (int j = 0; j < 4; ++j) acc[j] = WMMA_BF16(a, bfr[j], acc[j]);

  const int n = lane & 15, g = (lane >> 4) & 1;
#pragma unroll
  for (int j = 0; j < 4; ++j) {
    const float bj = bo[n0 + 16 * j + n];
#pragma unroll
    for (int r = 0; r < 8; ++r) {
      const int t = m0 + r + 8 * g;
      out[(size_t)t * DD + n0 + 16 * j + n] = acc[j][r] + bj;
    }
  }
}

// ---------------------------------------------------------------------------
// Host-side launch
// Input order: x, mask, Wq, bq, Wk, bk, Wv, bv, Wo, bo
// mask is a jax bool array -> 1 byte per element on device (uint8 0/1).
// ---------------------------------------------------------------------------
extern "C" void kernel_launch(void* const* d_in, const int* in_sizes, int n_in,
                              void* d_out, int out_size, void* d_ws,
                              size_t ws_size, hipStream_t stream) {
  const float* x  = (const float*)d_in[0];
  const unsigned char* mask = (const unsigned char*)d_in[1];
  const float* Wq = (const float*)d_in[2];
  const float* bq = (const float*)d_in[3];
  const float* Wk = (const float*)d_in[4];
  const float* bk = (const float*)d_in[5];
  const float* Wv = (const float*)d_in[6];
  const float* bv = (const float*)d_in[7];
  const float* Wo = (const float*)d_in[8];
  const float* bo = (const float*)d_in[9];
  float* out = (float*)d_out;

  // workspace layout (bf16 intermediates), ~68 MB total
  char* ws = (char*)d_ws;
  const size_t XB_SZ = (size_t)MM * DD * sizeof(bf16);   // 12.6 MB
  const size_t W_SZ  = (size_t)DD * DD * sizeof(bf16);   // 1.18 MB
  bf16* xb  = (bf16*)(ws);
  bf16* wqb = (bf16*)(ws + XB_SZ);
  bf16* wkb = (bf16*)(ws + XB_SZ + 1 * W_SZ);
  bf16* wvb = (bf16*)(ws + XB_SZ + 2 * W_SZ);
  bf16* wob = (bf16*)(ws + XB_SZ + 3 * W_SZ);
  bf16* Qd  = (bf16*)(ws + XB_SZ + 4 * W_SZ);
  bf16* Kd  = (bf16*)(ws + 2 * XB_SZ + 4 * W_SZ);
  bf16* Vtd = (bf16*)(ws + 3 * XB_SZ + 4 * W_SZ);
  bf16* AOd = (bf16*)(ws + 4 * XB_SZ + 4 * W_SZ);

  // 1) fp32 -> bf16 converts
  {
    const int nx4 = MM * DD / 4;
    const int nw4 = DD * DD / 4;
    f32_to_bf16_kernel<<<dim3((nx4 + 255) / 256), dim3(256), 0, stream>>>(x, xb, nx4);
    f32_to_bf16_kernel<<<dim3((nw4 + 255) / 256), dim3(256), 0, stream>>>(Wq, wqb, nw4);
    f32_to_bf16_kernel<<<dim3((nw4 + 255) / 256), dim3(256), 0, stream>>>(Wk, wkb, nw4);
    f32_to_bf16_kernel<<<dim3((nw4 + 255) / 256), dim3(256), 0, stream>>>(Wv, wvb, nw4);
    f32_to_bf16_kernel<<<dim3((nw4 + 255) / 256), dim3(256), 0, stream>>>(Wo, wob, nw4);
  }

  // 2) QKV projections (scale folded into Q, mask applied, V transposed)
  qkv_proj_kernel<<<dim3(MM / 64, HH, 3), dim3(128), 0, stream>>>(
      xb, wqb, wkb, wvb, bq, bk, bv, mask, Qd, Kd, Vtd);

  // 3) fused flash attention -> AO bf16 [M, D]
  attn_kernel<<<dim3(SS / 128, HH, BB), dim3(256), 0, stream>>>(
      Qd, Kd, Vtd, mask, AOd);

  // 4) output projection -> fp32 d_out
  out_proj_kernel<<<dim3(MM / 64, DD / 64), dim3(128), 0, stream>>>(
      AOd, wob, bo, out);
}
